// GPTQLinear_3753801416786
// MI455X (gfx1250) — matrix operand order repaired and checked by reading, not compiled, not run on hardware
//
#include <hip/hip_runtime.h>

typedef __attribute__((ext_vector_type(16))) _Float16 v16h;
typedef __attribute__((ext_vector_type(2)))  _Float16 h2;
typedef __attribute__((ext_vector_type(8)))  float    v8f;

#define BM 128
#define BN 128
#define BK 64
#define LDA 72   // halves per A row (64 + 8 pad) -> 144B, 16B aligned rows
#define LDB 72
#define THREADS 256

union H4   { _Float16 h[4]; uint2 v; };
union Frag { uint4 u[2];    v16h  h; };
union U32H2 { unsigned int u; h2 h; };
union U4   { unsigned int a[4]; uint4 v; };

__global__ __launch_bounds__(THREADS, 1) void gptq_wmma_kernel(
    const float* __restrict__ x,        // [M, K] f32
    const int*   __restrict__ qw,       // [N, K/2] one byte (2 nibbles) per int32
    const float* __restrict__ sc,       // [N, G]
    const int*   __restrict__ zr,       // [N, G]
    const float* __restrict__ bias,     // [N]
    float*       __restrict__ out,      // [M, N]
    int M, int N, int K, int G)
{
  __shared__ _Float16 lds_a[BM * LDA];
  __shared__ _Float16 lds_b[BN * LDB];

  const int tid   = threadIdx.x;
  const int lane  = tid & 31;
  const int wave  = tid >> 5;
  const int waveM = wave & 3;     // 4 waves along M -> 32 rows each
  const int waveN = wave >> 2;    // 2 waves along N -> 64 cols each
  const int lr    = lane & 15;
  const int hs    = lane >> 4;

  const int n0 = blockIdx.x * BN;
  const int m0 = blockIdx.y * BM;

  const int halfK = K >> 1;
  const int gsize = K / G;

  // A loader: 128x64 f32 = 2048 float4, 8 per thread (16 lanes cover one row)
  const int a_r0 = tid >> 4;           // + 16*i
  const int a_c4 = tid & 15;           // float4 column
  // B loader: 128 rows x 32 int32, two threads per row, 16 int32 each
  const int b_row = tid >> 1;
  const int b_ib  = (tid & 1) << 4;    // int32 offset within chunk: 0 or 16

  float4 areg[8];
  int4   breg[4];
  h2     s2   = {(_Float16)0.f, (_Float16)0.f};   // packed scale
  h2     za2  = {(_Float16)0.f, (_Float16)0.f};   // packed (1024 + zero)

  v8f acc[2][4];
  #pragma unroll
  for (int i = 0; i < 2; ++i)
    #pragma unroll
    for (int j = 0; j < 4; ++j)
      #pragma unroll
      for (int e = 0; e < 8; ++e)
        acc[i][j][e] = 0.0f;

  auto load_global = [&](int kc) {
    const int kb = kc * BK;
    const float* xp = x + (size_t)(m0 + a_r0) * K + kb + a_c4 * 4;
    #pragma unroll
    for (int i = 0; i < 8; ++i)
      areg[i] = *(const float4*)(xp + (size_t)i * 16 * K);
    const int* qp = qw + (size_t)(n0 + b_row) * halfK + (kb >> 1) + b_ib;
    #pragma unroll
    for (int j = 0; j < 4; ++j)
      breg[j] = *(const int4*)(qp + j * 4);
    const int g = kb / gsize;   // BK=64 < group 128 -> one group per chunk
    const float s = sc[(size_t)(n0 + b_row) * G + g];
    const int   z = zr[(size_t)(n0 + b_row) * G + g];
    const _Float16 sh = (_Float16)s;
    const _Float16 zh = (_Float16)(1024.0f + (float)z);  // exact: integer < 2048
    s2[0] = sh;  s2[1] = sh;
    za2[0] = zh; za2[1] = zh;
  };

  auto store_lds = [&]() {
    #pragma unroll
    for (int i = 0; i < 8; ++i) {
      H4 u;
      u.h[0] = (_Float16)areg[i].x;
      u.h[1] = (_Float16)areg[i].y;
      u.h[2] = (_Float16)areg[i].z;
      u.h[3] = (_Float16)areg[i].w;
      *(uint2*)&lds_a[(a_r0 + 16 * i) * LDA + a_c4 * 4] = u.v;
    }
    #pragma unroll
    for (int j = 0; j < 4; ++j) {
      const unsigned int qv[4] = {(unsigned int)breg[j].x, (unsigned int)breg[j].y,
                                  (unsigned int)breg[j].z, (unsigned int)breg[j].w};
      U4 o;
      #pragma unroll
      for (int e = 0; e < 4; ++e) {
        // f16 mantissa stuffing: 0x6400|q == 1024.0h + q exactly (q in 0..15)
        U32H2 t;
        t.u = 0x64006400u | (qv[e] & 0xFu) | ((qv[e] & 0xF0u) << 12);
        U32H2 r;
        r.h = (t.h - za2) * s2;   // exact subtract, one packed mul
        o.a[e] = r.u;
      }
      *(uint4*)&lds_b[b_row * LDB + (b_ib << 1) + j * 8] = o.v;
    }
  };

  auto compute = [&]() {
    #pragma unroll
    for (int kk = 0; kk < BK; kk += 32) {
      v16h af[2], bf[4];
      #pragma unroll
      for (int i = 0; i < 2; ++i) {
        const int row = waveM * 32 + i * 16 + lr;
        Frag f;
        f.u[0] = *(const uint4*)&lds_a[row * LDA + kk + hs * 8];       // K 0..7 / 8..15
        f.u[1] = *(const uint4*)&lds_a[row * LDA + kk + 16 + hs * 8];  // K 16..23 / 24..31
        af[i] = f.h;
      }
      #pragma unroll
      for (int j = 0; j < 4; ++j) {
        const int nr = waveN * 64 + j * 16 + lr;
        const uint4* p = (const uint4*)&lds_b[nr * LDB + kk + hs * 8];
        Frag f;
        f.u[0] = p[0];
        f.u[1] = p[2];
        bf[j] = f.h;
      }
      #pragma unroll
      for (int i = 0; i < 2; ++i)
        #pragma unroll
        for (int j = 0; j < 4; ++j)
          acc[i][j] = __builtin_amdgcn_wmma_f32_16x16x32_f16(
              false, af[i], false, bf[j], (short)0, acc[i][j], false, false);
    }
  };

  const int NKC = K / BK;
  load_global(0);
  for (int kc = 0; kc < NKC; ++kc) {
    store_lds();
    __syncthreads();
    if (kc + 1 < NKC) load_global(kc + 1);   // overlap VMEM with WMMA
    compute();
    __syncthreads();
  }

  // Epilogue: C/D layout -> VGPR v holds (m = v + hs*8, n = lane&15)
  #pragma unroll
  for (int j = 0; j < 4; ++j) {
    const int ng = n0 + waveN * 64 + j * 16 + lr;
    const float bv = bias[ng];
    #pragma unroll
    for (int i = 0; i < 2; ++i) {
      const int mb = m0 + waveM * 32 + i * 16 + hs * 8;
      #pragma unroll
      for (int v = 0; v < 8; ++v)
        out[(size_t)(mb + v) * N + ng] = acc[i][j][v] + bv;
    }
  }
}

extern "C" void kernel_launch(void* const* d_in, const int* in_sizes, int n_in,
                              void* d_out, int out_size, void* d_ws, size_t ws_size,
                              hipStream_t stream) {
  const float* x  = (const float*)d_in[0];
  const int*   qw = (const int*)d_in[1];
  const float* sc = (const float*)d_in[2];
  const int*   zr = (const int*)d_in[3];
  const float* bs = (const float*)d_in[4];
  float* out = (float*)d_out;

  const int OUT = in_sizes[4];
  const int IN  = 2 * (in_sizes[1] / OUT);
  const int M   = in_sizes[0] / IN;
  const int G   = in_sizes[2] / OUT;

  dim3 grid(OUT / BN, M / BM);   // 32 x 64 = 2048 workgroups
  gptq_wmma_kernel<<<grid, THREADS, 0, stream>>>(x, qw, sc, zr, bs, out, M, OUT, IN, G);
}
